// MultiHeadAttention_20186346291773
// MI455X (gfx1250) — matrix operand order repaired and checked by reading, not compiled, not run on hardware
//
#include <hip/hip_runtime.h>
#include <hip/hip_bf16.h>

// ---------------- problem constants ----------------
constexpr int BATCH = 2;
constexpr int SEQ   = 2048;
constexpr int DMOD  = 1024;
constexpr int NH    = 16;
constexpr int DH    = 64;           // DMOD / NH
constexpr float INV_SCALE = 0.125f; // 1/sqrt(64)

typedef __bf16 bf16_t;
typedef __attribute__((ext_vector_type(4)))  bf16_t v4bf;
typedef __attribute__((ext_vector_type(8)))  bf16_t v8bf;
typedef __attribute__((ext_vector_type(16))) bf16_t v16bf;
typedef __attribute__((ext_vector_type(8)))  float  v8f;

// ---------------- WMMA helper ----------------
__device__ __forceinline__ v8f wmma_bf16(v16bf a, v16bf b, v8f c) {
    // (neg_a, A, neg_b, B, c_mod, C, reuse_a, reuse_b)
    return __builtin_amdgcn_wmma_f32_16x16x32_bf16(
        false, a, false, b, (short)0, c, false, false);
}

__device__ __forceinline__ v8f zero8() {
    return (v8f){0.f,0.f,0.f,0.f,0.f,0.f,0.f,0.f};
}

// ---------------- bf16 fragment loaders (no conversion, direct b128 loads) ----
// A fragment: 16x32 (MxK) bf16 row-major, leading dim ld (elements).
// row=(lane&15); contiguous 16-elem K chunk at (lane>>4)*16.
__device__ __forceinline__ v16bf frag_a_bf(const bf16_t* __restrict__ tile,
                                           int ld, int lane) {
    const int r  = lane & 15;
    const int kb = (lane >> 4) << 4;
    const bf16_t* p = tile + (size_t)r * ld + kb;
    v8bf lo = *(const v8bf*)p;          // 16B
    v8bf hi = *(const v8bf*)(p + 8);    // 16B
    return __builtin_shufflevector(lo, hi,
        0,1,2,3,4,5,6,7,8,9,10,11,12,13,14,15);
}

// B fragment: 32x16 (KxN) with B[k][n] = src[n][k] bf16 row-major.
// Lane holds n=(lane&15); contiguous 16-elem K chunk at (lane>>4)*16.
__device__ __forceinline__ v16bf frag_b_bf(const bf16_t* __restrict__ tile,
                                           int ld, int lane) {
    const int n    = lane & 15;
    const int kofs = (lane >> 4) << 4;
    const bf16_t* p = tile + (size_t)n * ld + kofs;
    v8bf lo = *(const v8bf*)p;
    v8bf hi = *(const v8bf*)(p + 8);
    return __builtin_shufflevector(lo, hi,
        0,1,2,3,4,5,6,7,8,9,10,11,12,13,14,15);
}

// B fragment from column-strided bf16 source (V with lane's head column fixed)
__device__ __forceinline__ v16bf frag_b_strided(const bf16_t* __restrict__ col0,
                                                int stride) {
    bf16_t t[16];
#pragma unroll
    for (int e = 0; e < 16; ++e) t[e] = col0[(size_t)e * stride];
    v16bf f;
#pragma unroll
    for (int e = 0; e < 16; ++e) f[e] = t[e];
    return f;
}

// ---------------- fp32 -> bf16 bulk convert ----------------
__global__ __launch_bounds__(256) void cvt_bf16_kernel(
    const float* __restrict__ src, bf16_t* __restrict__ dst, int n) {
    const int i = (blockIdx.x * 256 + threadIdx.x) * 4;
    if (i < n) {
        const float4 x = *(const float4*)(src + i);
        v4bf o;
        o[0] = (bf16_t)x.x; o[1] = (bf16_t)x.y;
        o[2] = (bf16_t)x.z; o[3] = (bf16_t)x.w;
        *(v4bf*)(dst + i) = o;
    }
}

// ---------------- GEMM: C[M,N] = X[M,K] @ W[N,K]^T + bias (bf16 operands) ----
// 256 threads (8 waves); wave tile 32x32; block tile 64(M) x 128(N).
// Software-pipelined: fragments for k+32 are loaded while WMMAs run on k.
template <bool OUT_BF16>
__global__ __launch_bounds__(256) void gemm_bf16_kernel(
    const bf16_t* __restrict__ X, const bf16_t* __restrict__ W,
    const float* __restrict__ bias, void* __restrict__ Cout,
    int M, int N, int K)
{
    const int lane = threadIdx.x & 31;
    const int wave = threadIdx.x >> 5;
    const int wm = wave & 1;
    const int wn = wave >> 1;
    const int m0 = blockIdx.x * 64 + wm * 32;
    const int n0 = blockIdx.y * 128 + wn * 32;

    v8f acc[2][2];
#pragma unroll
    for (int i = 0; i < 2; ++i)
#pragma unroll
        for (int j = 0; j < 2; ++j) acc[i][j] = zero8();

    v16bf a_cur[2], b_cur[2];
#pragma unroll
    for (int sm = 0; sm < 2; ++sm)
        a_cur[sm] = frag_a_bf(X + (size_t)(m0 + sm * 16) * K, K, lane);
#pragma unroll
    for (int sn = 0; sn < 2; ++sn)
        b_cur[sn] = frag_b_bf(W + (size_t)(n0 + sn * 16) * K, K, lane);

    for (int kk = 0; kk < K; kk += 32) {
        const bool more = (kk + 32 < K);
        v16bf a_nxt[2], b_nxt[2];
        if (more) {
#pragma unroll
            for (int sm = 0; sm < 2; ++sm)
                a_nxt[sm] = frag_a_bf(X + (size_t)(m0 + sm * 16) * K + kk + 32, K, lane);
#pragma unroll
            for (int sn = 0; sn < 2; ++sn)
                b_nxt[sn] = frag_b_bf(W + (size_t)(n0 + sn * 16) * K + kk + 32, K, lane);
        }
#pragma unroll
        for (int sm = 0; sm < 2; ++sm)
#pragma unroll
            for (int sn = 0; sn < 2; ++sn)
                acc[sm][sn] = wmma_bf16(a_cur[sm], b_cur[sn], acc[sm][sn]);
        if (more) {
#pragma unroll
            for (int sm = 0; sm < 2; ++sm) a_cur[sm] = a_nxt[sm];
#pragma unroll
            for (int sn = 0; sn < 2; ++sn) b_cur[sn] = b_nxt[sn];
        }
    }

    const int cm = (lane >> 4) << 3;
    const int cn = lane & 15;
#pragma unroll
    for (int sm = 0; sm < 2; ++sm)
#pragma unroll
        for (int sn = 0; sn < 2; ++sn) {
            const int col = n0 + sn * 16 + cn;
            const float bb = bias[col];
#pragma unroll
            for (int r = 0; r < 8; ++r) {
                const size_t idx = (size_t)(m0 + sm * 16 + r + cm) * N + col;
                const float v = acc[sm][sn][r] + bb;
                if (OUT_BF16) ((bf16_t*)Cout)[idx] = (bf16_t)v;
                else          ((float*)Cout)[idx]  = v;
            }
        }
}

// ---------------- fused attention ----------------
// One block (8 waves) per (b, h, 16-query-row slab).
// Pass 1: QK^T WMMA tiles (bf16 operands), online row max/sum via width-16
//         shuffles, cross-wave merge in LDS (recompute > spill for a 128KB slab).
// Pass 2: recompute QK^T, write softmax probabilities (fp32) to d_out once,
//         stage each 16x32 P-tile as bf16 in per-wave LDS to re-layout
//         C-tile -> A-fragment, accumulate P@V with WMMA, reduce across waves.
__global__ __launch_bounds__(256) void fused_attn_kernel(
    const bf16_t* __restrict__ Q, const bf16_t* __restrict__ Kp,
    const bf16_t* __restrict__ V, const int* __restrict__ mask,
    float* __restrict__ scores, bf16_t* __restrict__ concat)
{
    __shared__ float  red_m[8][16];
    __shared__ float  red_s[8][16];
    __shared__ float  m_fin[16];
    __shared__ float  s_inv[16];
    __shared__ bf16_t ptile[8][16 * 32];   // per-wave P staging (row-major, ld=32)
    __shared__ float  attn_part[8][16 * 64];

    const int lane = threadIdx.x & 31;
    const int wave = threadIdx.x >> 5;
    const int g   = blockIdx.x;            // BATCH * NH * (SEQ/16)
    const int b   = g >> 11;               // NH*(SEQ/16) = 2048
    const int rem = g & 2047;
    const int h   = rem >> 7;              // SEQ/16 = 128
    const int qbase = (rem & 127) << 4;

    const int cm = (lane >> 4) << 3;       // C-tile: +8 rows for upper half-wave
    const int cn = lane & 15;
    const int qtop = qbase + 15;

    // Q fragments for K-dim 0..31 and 32..63 (used by both passes)
    const bf16_t* qtile = Q + ((size_t)b * SEQ + qbase) * DMOD + h * DH;
    v16bf qa[2];
    qa[0] = frag_a_bf(qtile, DMOD, lane);
    qa[1] = frag_a_bf(qtile + 32, DMOD, lane);
    const bf16_t* kbasep = Kp + (size_t)b * SEQ * DMOD + h * DH;

    // ---------------- pass 1: online max / sumexp ----------------
    float m_run[8], s_run[8];
#pragma unroll
    for (int r = 0; r < 8; ++r) { m_run[r] = -3.0e38f; s_run[r] = 0.f; }

    for (int kb = wave * 32; kb <= qtop; kb += 256) {
        v16bf b0a = frag_b_bf(kbasep + (size_t)kb * DMOD,             DMOD, lane);
        v16bf b0b = frag_b_bf(kbasep + (size_t)kb * DMOD + 32,        DMOD, lane);
        v16bf b1a = frag_b_bf(kbasep + (size_t)(kb + 16) * DMOD,      DMOD, lane);
        v16bf b1b = frag_b_bf(kbasep + (size_t)(kb + 16) * DMOD + 32, DMOD, lane);
        v8f c0 = zero8(), c1 = zero8();
        c0 = wmma_bf16(qa[0], b0a, c0);
        c0 = wmma_bf16(qa[1], b0b, c0);
        c1 = wmma_bf16(qa[0], b1a, c1);
        c1 = wmma_bf16(qa[1], b1b, c1);

        const int k0 = kb + cn, k1 = kb + 16 + cn;
#pragma unroll
        for (int r = 0; r < 8; ++r) {
            const int q = qbase + cm + r;
            const float v0 = (k0 <= q) ? c0[r] * INV_SCALE : -1e9f;
            const float v1 = (k1 <= q) ? c1[r] * INV_SCALE : -1e9f;
            float t = fmaxf(v0, v1);
#pragma unroll
            for (int off = 8; off > 0; off >>= 1)
                t = fmaxf(t, __shfl_xor(t, off, 16));
            const float mnew = fmaxf(m_run[r], t);
            float e = __expf(v0 - mnew) + __expf(v1 - mnew);
#pragma unroll
            for (int off = 8; off > 0; off >>= 1)
                e += __shfl_xor(e, off, 16);
            s_run[r] = s_run[r] * __expf(m_run[r] - mnew) + e;
            m_run[r] = mnew;
        }
    }
    if (cn == 0) {   // lanes 0 and 16 publish rows cm..cm+7
#pragma unroll
        for (int r = 0; r < 8; ++r) {
            red_m[wave][cm + r] = m_run[r];
            red_s[wave][cm + r] = s_run[r];
        }
    }
    __syncthreads();

    if (threadIdx.x < 16) {
        const int t = threadIdx.x;
        float m = -3.0e38f;
#pragma unroll
        for (int w = 0; w < 8; ++w) m = fmaxf(m, red_m[w][t]);
        float s = 0.f;
#pragma unroll
        for (int w = 0; w < 8; ++w) s += red_s[w][t] * __expf(red_m[w][t] - m);
        m_fin[t] = m;
        s_inv[t] = 1.f / s;
    }
    __syncthreads();

    float mf[8], is[8];
#pragma unroll
    for (int r = 0; r < 8; ++r) { mf[r] = m_fin[cm + r]; is[r] = s_inv[cm + r]; }

    // ---------------- pass 2: write probs + P@V ----------------
    v8f acc[4];
#pragma unroll
    for (int i = 0; i < 4; ++i) acc[i] = zero8();

    float*  srow = scores + (((size_t)b * NH + h) * SEQ + qbase) * SEQ;
    bf16_t* pt   = &ptile[wave][0];
    const int bn   = lane & 15;
    const int bkof = (lane >> 4) << 4;
    const bf16_t* vbase = V + (size_t)b * SEQ * DMOD + h * DH;

    for (int kb = wave * 32; kb < SEQ; kb += 256) {
        if (kb > qtop) {   // fully masked: zero-fill output slab
#pragma unroll
            for (int r = 0; r < 8; ++r) {
                srow[(size_t)(cm + r) * SEQ + kb + cn]      = 0.f;
                srow[(size_t)(cm + r) * SEQ + kb + 16 + cn] = 0.f;
            }
            continue;
        }
        v16bf b0a = frag_b_bf(kbasep + (size_t)kb * DMOD,             DMOD, lane);
        v16bf b0b = frag_b_bf(kbasep + (size_t)kb * DMOD + 32,        DMOD, lane);
        v16bf b1a = frag_b_bf(kbasep + (size_t)(kb + 16) * DMOD,      DMOD, lane);
        v16bf b1b = frag_b_bf(kbasep + (size_t)(kb + 16) * DMOD + 32, DMOD, lane);
        v8f c0 = zero8(), c1 = zero8();
        c0 = wmma_bf16(qa[0], b0a, c0);
        c0 = wmma_bf16(qa[1], b0b, c0);
        c1 = wmma_bf16(qa[0], b1a, c1);
        c1 = wmma_bf16(qa[1], b1b, c1);

        const int k0 = kb + cn, k1 = kb + 16 + cn;
#pragma unroll
        for (int r = 0; r < 8; ++r) {
            const int q = qbase + cm + r;
            float p0 = 0.f, p1 = 0.f;
            if (mask[(size_t)q * SEQ + k0] != 0)
                p0 = __expf(c0[r] * INV_SCALE - mf[r]) * is[r];
            if (mask[(size_t)q * SEQ + k1] != 0)
                p1 = __expf(c1[r] * INV_SCALE - mf[r]) * is[r];
            srow[(size_t)(cm + r) * SEQ + k0] = p0;     // fp32 output scores
            srow[(size_t)(cm + r) * SEQ + k1] = p1;
            pt[(cm + r) * 32 + cn]      = (bf16_t)p0;   // bf16 LDS staging
            pt[(cm + r) * 32 + 16 + cn] = (bf16_t)p1;
        }
        // same-wave LDS ops are in-order: re-read as A fragment (16x32, ld=32)
        {
            v16bf ap = frag_a_bf(pt, 32, lane);
#pragma unroll
            for (int nt = 0; nt < 4; ++nt) {
                v16bf vb = frag_b_strided(
                    vbase + (size_t)(kb + bkof) * DMOD + nt * 16 + bn, DMOD);
                acc[nt] = wmma_bf16(ap, vb, acc[nt]);
            }
        }
    }

    // ---------------- cross-wave reduction of P@V partials ----------------
    float* ap = &attn_part[wave][0];
#pragma unroll
    for (int nt = 0; nt < 4; ++nt)
#pragma unroll
        for (int r = 0; r < 8; ++r)
            ap[(cm + r) * 64 + nt * 16 + cn] = acc[nt][r];
    __syncthreads();

    for (int idx = threadIdx.x; idx < 16 * 64; idx += 256) {
        float s = 0.f;
#pragma unroll
        for (int w = 0; w < 8; ++w) s += attn_part[w][idx];
        const int row = idx >> 6;
        const int col = idx & 63;
        concat[((size_t)b * SEQ + qbase + row) * DMOD + h * DH + col] = (bf16_t)s;
    }
}

// ---------------- host launcher ----------------
extern "C" void kernel_launch(void* const* d_in, const int* in_sizes, int n_in,
                              void* d_out, int out_size, void* d_ws, size_t ws_size,
                              hipStream_t stream) {
    const float* query = (const float*)d_in[0];
    const float* key   = (const float*)d_in[1];
    const float* value = (const float*)d_in[2];
    const int*   mask  = (const int*)d_in[3];
    const float* Wq = (const float*)d_in[4];
    const float* bq = (const float*)d_in[5];
    const float* Wk = (const float*)d_in[6];
    const float* bk = (const float*)d_in[7];
    const float* Wv = (const float*)d_in[8];
    const float* bv = (const float*)d_in[9];
    const float* Wo = (const float*)d_in[10];
    const float* bo = (const float*)d_in[11];

    const size_t BSD = (size_t)BATCH * SEQ * DMOD;   // 4M elements
    const size_t DD  = (size_t)DMOD * DMOD;          // 1M elements
    float* out    = (float*)d_out;
    float* scores = out + BSD;                        // [B,H,S,S] fp32

    // bf16 workspace layout
    bf16_t* xq = (bf16_t*)d_ws;          // query  bf16 [B,S,D]
    bf16_t* xk = xq + BSD;
    bf16_t* xv = xk + BSD;
    bf16_t* wq = xv + BSD;               // weights bf16 [D,D]
    bf16_t* wk = wq + DD;
    bf16_t* wv = wk + DD;
    bf16_t* wo = wv + DD;
    bf16_t* qp = wo + DD;                // projections bf16 [B,S,D]
    bf16_t* kp = qp + BSD;
    bf16_t* vp = kp + BSD;
    bf16_t* cc = vp + BSD;               // concat bf16 [B,S,D]

    dim3 blk(256);
    const int M = BATCH * SEQ;                        // 4096
    dim3 gemm_grid(M / 64, DMOD / 128);               // 64 x 8
    const int cvt_bsd = (int)(BSD / 4 / 256);
    const int cvt_dd  = (int)(DD  / 4 / 256);

    // one-time bf16 conversion of activations and weights
    cvt_bf16_kernel<<<dim3(cvt_bsd), blk, 0, stream>>>(query, xq, (int)BSD);
    cvt_bf16_kernel<<<dim3(cvt_bsd), blk, 0, stream>>>(key,   xk, (int)BSD);
    cvt_bf16_kernel<<<dim3(cvt_bsd), blk, 0, stream>>>(value, xv, (int)BSD);
    cvt_bf16_kernel<<<dim3(cvt_dd),  blk, 0, stream>>>(Wq, wq, (int)DD);
    cvt_bf16_kernel<<<dim3(cvt_dd),  blk, 0, stream>>>(Wk, wk, (int)DD);
    cvt_bf16_kernel<<<dim3(cvt_dd),  blk, 0, stream>>>(Wv, wv, (int)DD);
    cvt_bf16_kernel<<<dim3(cvt_dd),  blk, 0, stream>>>(Wo, wo, (int)DD);

    // projections (bf16 in, bf16 out)
    gemm_bf16_kernel<true><<<gemm_grid, blk, 0, stream>>>(xq, wq, bq, qp, M, DMOD, DMOD);
    gemm_bf16_kernel<true><<<gemm_grid, blk, 0, stream>>>(xk, wk, bk, kp, M, DMOD, DMOD);
    gemm_bf16_kernel<true><<<gemm_grid, blk, 0, stream>>>(xv, wv, bv, vp, M, DMOD, DMOD);

    // fused softmax-attention (writes fp32 scores + bf16 concat)
    fused_attn_kernel<<<dim3(BATCH * NH * (SEQ / 16)), blk, 0, stream>>>(
        qp, kp, vp, mask, scores, cc);

    // output projection (bf16 in, fp32 out)
    gemm_bf16_kernel<false><<<gemm_grid, blk, 0, stream>>>(cc, wo, bo, out, M, DMOD, DMOD);
}
